// GCN_pyg_78898549227791
// MI455X (gfx1250) — matrix operand order repaired and checked by reading, not compiled, not run on hardware
//
#include <hip/hip_runtime.h>
#include <hip/hip_bf16.h>

#define N_NODES 50000
#define N_EDGES 800000
#define NFEAT   256
#define NHID    128
#define NTOPIC  64

typedef __attribute__((ext_vector_type(16))) __bf16 v16bf;
typedef __attribute__((ext_vector_type(8)))  float  v8f;

// ---------------------------------------------------------------------------
// Init: zero degree + layer-1 aggregation buffer, seed output with bias b2
// (so the final scatter's atomics land on top of the bias -> no epilogue pass)
// ---------------------------------------------------------------------------
__global__ __launch_bounds__(256) void init_kernel(float* __restrict__ deg,
                                                   float* __restrict__ agg,
                                                   float* __restrict__ out,
                                                   const float* __restrict__ b2) {
    int i = blockIdx.x * 256 + threadIdx.x;
    if (i < N_NODES * NHID)   agg[i] = 0.0f;
    if (i < N_NODES)          deg[i] = 0.0f;
    if (i < N_NODES * NTOPIC) out[i] = b2[i & (NTOPIC - 1)];
}

// ---------------------------------------------------------------------------
// In-degree via L2 float atomics
// ---------------------------------------------------------------------------
__global__ __launch_bounds__(256) void deg_kernel(const long long* __restrict__ dst,
                                                  float* __restrict__ deg) {
    int e = blockIdx.x * 256 + threadIdx.x;
    if (e >= N_EDGES) return;
    atomicAdd(&deg[(int)dst[e]], 1.0f);
}

__global__ __launch_bounds__(256) void dis_kernel(const float* __restrict__ deg,
                                                  float* __restrict__ dis) {
    int i = blockIdx.x * 256 + threadIdx.x;
    if (i >= N_NODES) return;
    float d = deg[i];
    dis[i] = (d > 0.0f) ? rsqrtf(fmaxf(d, 1.0f)) : 0.0f;
}

// ---------------------------------------------------------------------------
// GEMM1: H1[50000,128] = bf16(X[50000,256]) @ bf16(W1[256,128]), fp32 acc.
// One wave = one 16x16 output tile. 8 waves/block cover N=128 exactly;
// gridDim.x = 50000/16 = 3125 covers M exactly (EXEC all-ones for WMMA).
// ---------------------------------------------------------------------------
__global__ __launch_bounds__(256) void gemm1_kernel(const float* __restrict__ X,
                                                    const float* __restrict__ W,
                                                    float* __restrict__ H) {
    const int lane = threadIdx.x & 31;
    const int wv   = threadIdx.x >> 5;        // n-tile 0..7
    const int r    = lane & 15;
    const int hi   = lane >> 4;               // half-wave select
    const int row  = blockIdx.x * 16 + r;     // A row handled by this lane
    const int ncol = wv * 16 + r;             // B column handled by this lane

    v8f acc = {};
    for (int k0 = 0; k0 < NFEAT; k0 += 32) {
        v16bf a, b;
        const float* xp = X + (size_t)row * NFEAT + k0 + hi * 8;
        const float* wp = W + (size_t)(k0 + hi * 8) * NHID + ncol;
#pragma unroll
        for (int e = 0; e < 8; ++e)  a[e] = (__bf16)xp[e];          // K = kb+e
#pragma unroll
        for (int e = 8; e < 16; ++e) a[e] = (__bf16)xp[e + 8];      // K = kb+16+(e-8)
#pragma unroll
        for (int e = 0; e < 16; ++e) b[e] = (__bf16)wp[(size_t)(e < 8 ? e : e + 8) * NHID];
        acc = __builtin_amdgcn_wmma_f32_16x16x32_bf16(false, a, false, b,
                                                      (short)0, acc, false, false);
    }
    // D elem v -> row v + 8*hi, col = ncol&15 within tile
    float* hp = H + ((size_t)blockIdx.x * 16 + 8 * hi) * NHID + wv * 16 + r;
#pragma unroll
    for (int v = 0; v < 8; ++v) hp[(size_t)v * NHID] = acc[v];
}

// ---------------------------------------------------------------------------
// Edge scatter layer 1: AGG1[dst] += norm * H1[src]  (128 feats = 32 float4)
// ---------------------------------------------------------------------------
__global__ __launch_bounds__(256) void scatter1_kernel(const long long* __restrict__ src,
                                                       const long long* __restrict__ dst,
                                                       const float* __restrict__ dis,
                                                       const float* __restrict__ H,
                                                       float* __restrict__ AGG) {
    int idx = blockIdx.x * 256 + threadIdx.x;   // 25.6M threads
    int e = idx >> 5, c = idx & 31;
    if (e >= N_EDGES) return;
    int s = (int)src[e], d = (int)dst[e];
    float nm = dis[s] * dis[d];
    float4 v = ((const float4*)(H + (size_t)s * NHID))[c];
    float* op = AGG + (size_t)d * NHID + c * 4;
    atomicAdd(op + 0, nm * v.x);
    atomicAdd(op + 1, nm * v.y);
    atomicAdd(op + 2, nm * v.z);
    atomicAdd(op + 3, nm * v.w);
}

// ---------------------------------------------------------------------------
// GEMM2: H2[50000,64] = bf16(relu(AGG1 + b1)) @ bf16(W2[128,64]), fp32 acc.
// Bias + ReLU fused into the A-fragment load. 4 waves/block cover N=64.
// ---------------------------------------------------------------------------
__global__ __launch_bounds__(128) void gemm2_kernel(const float* __restrict__ AGG,
                                                    const float* __restrict__ bias1,
                                                    const float* __restrict__ W,
                                                    float* __restrict__ H2) {
    const int lane = threadIdx.x & 31;
    const int wv   = threadIdx.x >> 5;        // n-tile 0..3
    const int r    = lane & 15;
    const int hi   = lane >> 4;
    const int row  = blockIdx.x * 16 + r;
    const int ncol = wv * 16 + r;

    v8f acc = {};
    for (int k0 = 0; k0 < NHID; k0 += 32) {
        v16bf a, b;
        const float* ap = AGG   + (size_t)row * NHID + k0 + hi * 8;
        const float* bp = bias1 + k0 + hi * 8;
        const float* wp = W + (size_t)(k0 + hi * 8) * NTOPIC + ncol;
#pragma unroll
        for (int e = 0; e < 8; ++e) {
            float v = ap[e] + bp[e];
            a[e] = (__bf16)fmaxf(v, 0.0f);
        }
#pragma unroll
        for (int e = 8; e < 16; ++e) {
            float v = ap[e + 8] + bp[e + 8];
            a[e] = (__bf16)fmaxf(v, 0.0f);
        }
#pragma unroll
        for (int e = 0; e < 16; ++e) b[e] = (__bf16)wp[(size_t)(e < 8 ? e : e + 8) * NTOPIC];
        acc = __builtin_amdgcn_wmma_f32_16x16x32_bf16(false, a, false, b,
                                                      (short)0, acc, false, false);
    }
    float* hp = H2 + ((size_t)blockIdx.x * 16 + 8 * hi) * NTOPIC + wv * 16 + r;
#pragma unroll
    for (int v = 0; v < 8; ++v) hp[(size_t)v * NTOPIC] = acc[v];
}

// ---------------------------------------------------------------------------
// Edge scatter layer 2: OUT[dst] += norm * H2[src]  (64 feats = 16 float4)
// OUT was pre-seeded with b2 by init_kernel.
// ---------------------------------------------------------------------------
__global__ __launch_bounds__(256) void scatter2_kernel(const long long* __restrict__ src,
                                                       const long long* __restrict__ dst,
                                                       const float* __restrict__ dis,
                                                       const float* __restrict__ H2,
                                                       float* __restrict__ out) {
    int idx = blockIdx.x * 256 + threadIdx.x;   // 12.8M threads
    int e = idx >> 4, c = idx & 15;
    if (e >= N_EDGES) return;
    int s = (int)src[e], d = (int)dst[e];
    float nm = dis[s] * dis[d];
    float4 v = ((const float4*)(H2 + (size_t)s * NTOPIC))[c];
    float* op = out + (size_t)d * NTOPIC + c * 4;
    atomicAdd(op + 0, nm * v.x);
    atomicAdd(op + 1, nm * v.y);
    atomicAdd(op + 2, nm * v.z);
    atomicAdd(op + 3, nm * v.w);
}

extern "C" void kernel_launch(void* const* d_in, const int* in_sizes, int n_in,
                              void* d_out, int out_size, void* d_ws, size_t ws_size,
                              hipStream_t stream) {
    const float*     x   = (const float*)d_in[0];
    const long long* ei  = (const long long*)d_in[1];   // int64 [2, E]
    const float*     W1  = (const float*)d_in[2];
    const float*     b1  = (const float*)d_in[3];
    const float*     W2  = (const float*)d_in[4];
    const float*     b2  = (const float*)d_in[5];
    float*           out = (float*)d_out;

    const long long* src = ei;
    const long long* dst = ei + N_EDGES;

    // Workspace layout (floats): deg | dis | H1 (reused as H2) | AGG1
    float* ws  = (float*)d_ws;
    float* deg = ws;                         // 50000
    float* dis = ws + 51200;                 // 50000
    float* H1  = ws + 102400;                // 50000*128 = 6.4M
    float* AGG = H1 + (size_t)N_NODES * NHID;// 50000*128 = 6.4M
    float* H2  = H1;                         // H1 dead after scatter1 -> reuse

    init_kernel<<<(N_NODES * NHID + 255) / 256, 256, 0, stream>>>(deg, AGG, out, b2);
    deg_kernel<<<(N_EDGES + 255) / 256, 256, 0, stream>>>(dst, deg);
    dis_kernel<<<(N_NODES + 255) / 256, 256, 0, stream>>>(deg, dis);
    gemm1_kernel<<<N_NODES / 16, 256, 0, stream>>>(x, W1, H1);
    scatter1_kernel<<<(N_EDGES * 32) / 256, 256, 0, stream>>>(src, dst, dis, H1, AGG);
    gemm2_kernel<<<N_NODES / 16, 128, 0, stream>>>(AGG, b1, W2, H2);
    scatter2_kernel<<<(N_EDGES * 16) / 256, 256, 0, stream>>>(src, dst, dis, H2, out);
}
